// BERT1E_4947802325048
// MI455X (gfx1250) — matrix operand order repaired and checked by reading, not compiled, not run on hardware
//
#include <hip/hip_runtime.h>

// Problem constants from the reference
#define Bsz  32
#define Ssz  512
#define Dsz  768
#define Psz  256
#define Hsz  12
#define DHsz 64
#define DFFsz 3072

typedef __attribute__((ext_vector_type(16))) _Float16 v16h;
typedef __attribute__((ext_vector_type(8)))  _Float16 v8h;
typedef __attribute__((ext_vector_type(2)))  _Float16 v2h;
typedef __attribute__((ext_vector_type(8)))  float    v8f;

// ---------------------------------------------------------------------------
// f32 -> f16 conversion (weights), 2 elements per thread, packed b32 store
// n must be even (all weight matrices are).
// ---------------------------------------------------------------------------
__global__ __launch_bounds__(256) void cvt_f16_kernel(const float* __restrict__ in,
                                                      _Float16* __restrict__ out, int n) {
    int i = (blockIdx.x * blockDim.x + threadIdx.x) * 2;
    if (i < n) {
        v2h h;
        h.x = (_Float16)in[i];
        h.y = (_Float16)in[i + 1];
        *(v2h*)(out + i) = h;
    }
}

// ---------------------------------------------------------------------------
// Gather + ragged mean-pool. Produces:
//   s_emb  f32 [P,D]           (source phrase, for loss + memory slot 1)
//   pool_q f32 [2P,D]          (rows 0..P-1 = p_emb, P..2P-1 = n_emb; residual)
//   qx_h   f16 [2P,D]          (same, GEMM input)
//   mem_h  f16 [2P,D]          (row 2p = cls, row 2p+1 = s_emb; K/V GEMM input)
// ---------------------------------------------------------------------------
__global__ __launch_bounds__(256) void pool_kernel(
    const float* __restrict__ sent, const int* __restrict__ sidx,
    const int* __restrict__ ss, const int* __restrict__ ps, const int* __restrict__ ns,
    float* __restrict__ s_emb, float* __restrict__ pool_q,
    _Float16* __restrict__ qx_h, _Float16* __restrict__ mem_h) {
    int p = blockIdx.x;
    const float* g = sent + (size_t)sidx[p] * Ssz * Dsz;
    int s0 = ss[2*p], s1 = ss[2*p+1];
    int p0 = ps[2*p], p1 = ps[2*p+1];
    int n0 = ns[2*p], n1 = ns[2*p+1];
    float is = 1.0f / (float)(s1 - s0);
    float ip = 1.0f / (float)(p1 - p0);
    float in_ = 1.0f / (float)(n1 - n0);
    for (int d = threadIdx.x; d < Dsz; d += blockDim.x) {
        float as = 0.f, ap = 0.f, an = 0.f;
        for (int t = s0; t < s1; ++t) as += g[t*Dsz + d];
        for (int t = p0; t < p1; ++t) ap += g[t*Dsz + d];
        for (int t = n0; t < n1; ++t) an += g[t*Dsz + d];
        float vs = as*is, vp = ap*ip, vn = an*in_;
        float cls = g[d];
        s_emb[(size_t)p*Dsz + d] = vs;
        pool_q[(size_t)p*Dsz + d] = vp;
        pool_q[(size_t)(Psz+p)*Dsz + d] = vn;
        qx_h[(size_t)p*Dsz + d] = (_Float16)vp;
        qx_h[(size_t)(Psz+p)*Dsz + d] = (_Float16)vn;
        mem_h[(size_t)(2*p)*Dsz + d] = (_Float16)cls;
        mem_h[(size_t)(2*p+1)*Dsz + d] = (_Float16)vs;
    }
}

// ---------------------------------------------------------------------------
// f16 WMMA GEMM, fully shape-specialized: C[M,N] = act(A[M,K] @ B[K,N] + bias)
// Block = 256 threads = 8 waves, block tile 128x64, wave tile 32x32
// (2x2 grid of 16x16x32 WMMA tiles), K-loop step 32.
// N, K, RELU, OUTF16 are compile-time so the epilogue stores use constant
// instruction offsets (no runtime 64-bit address chains).
// ---------------------------------------------------------------------------
template <int N, int K, int RELU, int OUTF16>
__global__ __launch_bounds__(256) void gemm_wmma_kernel(
    const _Float16* __restrict__ A, const _Float16* __restrict__ B,
    const float* __restrict__ bias, void* __restrict__ Cout) {
    const int tid  = threadIdx.x;
    const int wave = tid >> 5;
    const int lane = tid & 31;
    const int lm   = lane & 15;
    const int hi   = lane >> 4;

    const int brow = blockIdx.x * 128 + (wave & 3) * 32;  // wave's 32-row strip
    const int bcol = blockIdx.y * 64  + (wave >> 2) * 32; // wave's 32-col strip

    v8f acc[2][2] = {};

    const _Float16* ap0 = A + (size_t)(brow + lm) * K + hi * 8;        // mt=0
    const _Float16* ap1 = ap0 + (size_t)16 * K;                        // mt=1
    const _Float16* bp  = B + (size_t)(hi * 8) * N + bcol + lm;        // col lm, k half hi

    for (int k0 = 0; k0 < K; k0 += 32) {
        v16h afrag[2], bfrag[2];
        {
            v8h lo  = *(const v8h*)(ap0 + k0);
            v8h hi8 = *(const v8h*)(ap0 + k0 + 16);
            afrag[0] = __builtin_shufflevector(lo, hi8,
                0,1,2,3,4,5,6,7,8,9,10,11,12,13,14,15);
            lo  = *(const v8h*)(ap1 + k0);
            hi8 = *(const v8h*)(ap1 + k0 + 16);
            afrag[1] = __builtin_shufflevector(lo, hi8,
                0,1,2,3,4,5,6,7,8,9,10,11,12,13,14,15);
        }
        {
            const _Float16* bk = bp + (size_t)k0 * N;
            #pragma unroll
            for (int nt = 0; nt < 2; ++nt)
                #pragma unroll
                for (int s = 0; s < 16; ++s)
                    bfrag[nt][s] = bk[(size_t)(s < 8 ? s : s + 8) * N + nt * 16];
        }
        #pragma unroll
        for (int mt = 0; mt < 2; ++mt)
            #pragma unroll
            for (int nt = 0; nt < 2; ++nt)
                acc[mt][nt] = __builtin_amdgcn_wmma_f32_16x16x32_f16(
                    false, afrag[mt], false, bfrag[nt],
                    (short)0, acc[mt][nt], false, false);
    }

    // Epilogue: bias + optional ReLU; constant-offset store runs.
    #pragma unroll
    for (int mt = 0; mt < 2; ++mt) {
        const int rbase = brow + mt*16 + hi*8;
        #pragma unroll
        for (int nt = 0; nt < 2; ++nt) {
            const int col = bcol + nt*16 + lm;
            const float bb = bias[col];
            const size_t base = (size_t)rbase * N + col;
            #pragma unroll
            for (int r = 0; r < 8; ++r) {
                float v = acc[mt][nt][r] + bb;
                if (RELU) v = v > 0.0f ? v : 0.0f;
                if (OUTF16) ((_Float16*)Cout)[base + (size_t)r * N] = (_Float16)v;
                else        ((float*)Cout)[base + (size_t)r * N] = v;
            }
        }
    }
}

// ---------------------------------------------------------------------------
// 1-query, 2-slot cross attention. One block per query row (512 rows),
// 12 waves = one wave per head; lane-parallel dot over DH=64 (2 elems/lane).
// ---------------------------------------------------------------------------
__global__ __launch_bounds__(384) void attn_kernel(
    const float* __restrict__ Q, const float* __restrict__ Kc,
    const float* __restrict__ V, _Float16* __restrict__ attn_h) {
    const int row  = blockIdx.x;       // 0..511 (p rows then n rows)
    const int pair = row & (Psz - 1);
    const int head = threadIdx.x >> 5; // 0..11
    const int lane = threadIdx.x & 31;

    const float* q  = Q  + (size_t)row * Dsz + head * DHsz;
    const float* k0 = Kc + (size_t)(2*pair) * Dsz + head * DHsz;
    const float* k1 = k0 + Dsz;
    const float* v0 = V  + (size_t)(2*pair) * Dsz + head * DHsz;
    const float* v1 = v0 + Dsz;

    float qa = q[lane], qb = q[lane + 32];
    float s0 = qa * k0[lane] + qb * k0[lane + 32];
    float s1 = qa * k1[lane] + qb * k1[lane + 32];
    #pragma unroll
    for (int off = 16; off > 0; off >>= 1) {
        s0 += __shfl_xor(s0, off, 32);
        s1 += __shfl_xor(s1, off, 32);
    }
    s0 *= 0.125f; s1 *= 0.125f;           // 1/sqrt(64)
    float mx = fmaxf(s0, s1);
    float e0 = __expf(s0 - mx), e1 = __expf(s1 - mx);
    float inv = 1.0f / (e0 + e1);
    float a0 = e0 * inv, a1 = e1 * inv;

    _Float16* o = attn_h + (size_t)row * Dsz + head * DHsz;
    o[lane]      = (_Float16)(a0 * v0[lane]      + a1 * v1[lane]);
    o[lane + 32] = (_Float16)(a0 * v0[lane + 32] + a1 * v1[lane + 32]);
}

// ---------------------------------------------------------------------------
// y = LayerNorm(a + b) * gamma + beta  per row of 768; f32 + optional f16 out
// ---------------------------------------------------------------------------
template <int WANT_F16>
__global__ __launch_bounds__(256) void ln_kernel(
    const float* __restrict__ a, const float* __restrict__ b,
    const float* __restrict__ gamma, const float* __restrict__ beta,
    float* __restrict__ yf, _Float16* __restrict__ yh) {
    __shared__ float red[16];
    const int row = blockIdx.x;
    const int tid = threadIdx.x;
    const float* ar = a + (size_t)row * Dsz;
    const float* br = b + (size_t)row * Dsz;
    float x0 = ar[tid]       + br[tid];
    float x1 = ar[tid + 256] + br[tid + 256];
    float x2 = ar[tid + 512] + br[tid + 512];
    float s  = x0 + x1 + x2;
    float s2 = x0*x0 + x1*x1 + x2*x2;
    #pragma unroll
    for (int off = 16; off > 0; off >>= 1) {
        s  += __shfl_xor(s,  off, 32);
        s2 += __shfl_xor(s2, off, 32);
    }
    const int wave = tid >> 5, lane = tid & 31;
    if (lane == 0) { red[wave] = s; red[8 + wave] = s2; }
    __syncthreads();
    if (tid == 0) {
        float ts = 0.f, ts2 = 0.f;
        for (int w = 0; w < 8; ++w) { ts += red[w]; ts2 += red[8 + w]; }
        red[0] = ts; red[8] = ts2;
    }
    __syncthreads();
    const float mean = red[0] * (1.0f / Dsz);
    const float var  = red[8] * (1.0f / Dsz) - mean * mean;
    const float rs   = rsqrtf(var + 1e-5f);
    float xs[3] = {x0, x1, x2};
    #pragma unroll
    for (int j = 0; j < 3; ++j) {
        int d = tid + j * 256;
        float y = (xs[j] - mean) * rs * gamma[d] + beta[d];
        yf[(size_t)row * Dsz + d] = y;
        if (WANT_F16) yh[(size_t)row * Dsz + d] = (_Float16)y;
    }
}

// ---------------------------------------------------------------------------
// Triplet hinge loss: mean over P of max(||s-ap+eps|| - ||s-an+eps|| + 1, 0)
// One block of 256 threads, one pair per thread.
// ---------------------------------------------------------------------------
__global__ __launch_bounds__(256) void loss_kernel(
    const float* __restrict__ s_emb, const float* __restrict__ out,
    float* __restrict__ res) {
    __shared__ float red[8];
    const int p = threadIdx.x;
    const float* se = s_emb + (size_t)p * Dsz;
    const float* op = out + (size_t)p * Dsz;
    const float* on = out + (size_t)(Psz + p) * Dsz;
    float dp2 = 0.f, dn2 = 0.f;
    for (int d = 0; d < Dsz; ++d) {
        float ap = se[d] - op[d] + 1e-6f;
        float an = se[d] - on[d] + 1e-6f;
        dp2 += ap * ap;
        dn2 += an * an;
    }
    float v = sqrtf(dp2) - sqrtf(dn2) + 1.0f;
    v = v > 0.f ? v : 0.f;
    #pragma unroll
    for (int off = 16; off > 0; off >>= 1) v += __shfl_xor(v, off, 32);
    const int wave = p >> 5, lane = p & 31;
    if (lane == 0) red[wave] = v;
    __syncthreads();
    if (p == 0) {
        float t = 0.f;
        for (int w = 0; w < 8; ++w) t += red[w];
        res[0] = t * (1.0f / Psz);
    }
}

// ---------------------------------------------------------------------------
extern "C" void kernel_launch(void* const* d_in, const int* in_sizes, int n_in,
                              void* d_out, int out_size, void* d_ws, size_t ws_size,
                              hipStream_t stream) {
    const float* sent = (const float*)d_in[0];
    const int*   sidx = (const int*)d_in[1];
    const int*   ss   = (const int*)d_in[2];
    const int*   ps   = (const int*)d_in[3];
    const int*   ns   = (const int*)d_in[4];
    const float* Wq = (const float*)d_in[5];  const float* bq = (const float*)d_in[6];
    const float* Wk = (const float*)d_in[7];  const float* bk = (const float*)d_in[8];
    const float* Wv = (const float*)d_in[9];  const float* bv = (const float*)d_in[10];
    const float* Wo = (const float*)d_in[11]; const float* bo = (const float*)d_in[12];
    const float* ln1g = (const float*)d_in[13]; const float* ln1b = (const float*)d_in[14];
    const float* ln2g = (const float*)d_in[15]; const float* ln2b = (const float*)d_in[16];
    const float* W1 = (const float*)d_in[17]; const float* b1 = (const float*)d_in[18];
    const float* W2 = (const float*)d_in[19]; const float* b2 = (const float*)d_in[20];

    char* ws = (char*)d_ws;
    size_t off = 0;
    auto alloc = [&](size_t bytes) -> void* {
        void* p = ws + off;
        off = (off + bytes + 255) & ~(size_t)255;
        return p;
    };

    _Float16* Wq_h  = (_Float16*)alloc((size_t)Dsz * Dsz * 2);
    _Float16* Wk_h  = (_Float16*)alloc((size_t)Dsz * Dsz * 2);
    _Float16* Wv_h  = (_Float16*)alloc((size_t)Dsz * Dsz * 2);
    _Float16* Wo_h  = (_Float16*)alloc((size_t)Dsz * Dsz * 2);
    _Float16* W1_h  = (_Float16*)alloc((size_t)Dsz * DFFsz * 2);
    _Float16* W2_h  = (_Float16*)alloc((size_t)DFFsz * Dsz * 2);
    float*    s_emb = (float*)alloc((size_t)Psz * Dsz * 4);
    float*    poolq = (float*)alloc((size_t)2 * Psz * Dsz * 4);
    _Float16* qx_h  = (_Float16*)alloc((size_t)2 * Psz * Dsz * 2);
    _Float16* mem_h = (_Float16*)alloc((size_t)2 * Psz * Dsz * 2);
    float*    Qout  = (float*)alloc((size_t)2 * Psz * Dsz * 4);
    float*    Kout  = (float*)alloc((size_t)2 * Psz * Dsz * 4);
    float*    Vout  = (float*)alloc((size_t)2 * Psz * Dsz * 4);
    _Float16* att_h = (_Float16*)alloc((size_t)2 * Psz * Dsz * 2);
    float*    aproj = (float*)alloc((size_t)2 * Psz * Dsz * 4);
    float*    h_f   = (float*)alloc((size_t)2 * Psz * Dsz * 4);
    _Float16* h_h   = (_Float16*)alloc((size_t)2 * Psz * Dsz * 2);
    _Float16* f1_h  = (_Float16*)alloc((size_t)2 * Psz * DFFsz * 2);
    float*    f2_f  = (float*)alloc((size_t)2 * Psz * Dsz * 4);
    float*    out_f = (float*)alloc((size_t)2 * Psz * Dsz * 4);

    // 1) weights -> f16 (2 elems/thread)
    const int nW = Dsz * Dsz, nF = Dsz * DFFsz;
    cvt_f16_kernel<<<(nW/2 + 255) / 256, 256, 0, stream>>>(Wq, Wq_h, nW);
    cvt_f16_kernel<<<(nW/2 + 255) / 256, 256, 0, stream>>>(Wk, Wk_h, nW);
    cvt_f16_kernel<<<(nW/2 + 255) / 256, 256, 0, stream>>>(Wv, Wv_h, nW);
    cvt_f16_kernel<<<(nW/2 + 255) / 256, 256, 0, stream>>>(Wo, Wo_h, nW);
    cvt_f16_kernel<<<(nF/2 + 255) / 256, 256, 0, stream>>>(W1, W1_h, nF);
    cvt_f16_kernel<<<(nF/2 + 255) / 256, 256, 0, stream>>>(W2, W2_h, nF);

    // 2) gather + pool
    pool_kernel<<<Psz, 256, 0, stream>>>(sent, sidx, ss, ps, ns,
                                         s_emb, poolq, qx_h, mem_h);

    // 3) projections (M=512, N=768, K=768)
    dim3 gP(512 / 128, Dsz / 64);
    gemm_wmma_kernel<Dsz, Dsz, 0, 0><<<gP, 256, 0, stream>>>(qx_h,  Wq_h, bq, Qout);
    gemm_wmma_kernel<Dsz, Dsz, 0, 0><<<gP, 256, 0, stream>>>(mem_h, Wk_h, bk, Kout);
    gemm_wmma_kernel<Dsz, Dsz, 0, 0><<<gP, 256, 0, stream>>>(mem_h, Wv_h, bv, Vout);

    // 4) 2-slot softmax attention per head
    attn_kernel<<<512, 384, 0, stream>>>(Qout, Kout, Vout, att_h);

    // 5) output projection
    gemm_wmma_kernel<Dsz, Dsz, 0, 0><<<gP, 256, 0, stream>>>(att_h, Wo_h, bo, aproj);

    // 6) LN1(x + attn) -> f32 + f16
    ln_kernel<1><<<512, 256, 0, stream>>>(poolq, aproj, ln1g, ln1b, h_f, h_h);

    // 7) FFN1 (ReLU, f16-only output feeds FFN2)
    dim3 gF(512 / 128, DFFsz / 64);
    gemm_wmma_kernel<DFFsz, Dsz, 1, 1><<<gF, 256, 0, stream>>>(h_h, W1_h, b1, f1_h);

    // 8) FFN2 (K=3072)
    gemm_wmma_kernel<Dsz, DFFsz, 0, 0><<<gP, 256, 0, stream>>>(f1_h, W2_h, b2, f2_f);

    // 9) LN2(h + ffn) -> f32 only
    ln_kernel<0><<<512, 256, 0, stream>>>(h_f, f2_f, ln2g, ln2b, out_f, nullptr);

    // 10) triplet hinge loss -> scalar
    loss_kernel<<<1, 256, 0, stream>>>(s_emb, out_f, (float*)d_out);
}
